// CAM_50053548867817
// MI455X (gfx1250) — matrix operand order repaired and checked by reading, not compiled, not run on hardware
//
#include <hip/hip_runtime.h>
#include <math.h>

#define DEV __device__ __forceinline__

constexpr int BB = 8;      // batch
constexpr int NN = 2048;   // sequence length L
constexpr int DD = 1024;   // d_inner
constexpr int NS = 16;     // d_state
constexpr int RK = 32;     // dt_rank
constexpr int CC = 64;     // dt_rank + 2*d_state
constexpr int KC = 8;      // clusters

typedef __attribute__((ext_vector_type(16))) __bf16 v16bf;
typedef __attribute__((ext_vector_type(8)))  float  v8f;

struct alignas(16) U128 { unsigned int u[4]; };
union U16 { v16bf v; unsigned short s[16]; U128 q[2]; };

DEV unsigned short f2bf(float f) {
    union { float f; unsigned u; } x; x.f = f;
    unsigned r = x.u + 0x7fffu + ((x.u >> 16) & 1u);
    return (unsigned short)(r >> 16);
}
// fast softplus: single v_exp_f32 + v_log_f32 on the TRANS pipe
DEV float softplusf(float v) { return v > 20.f ? v : __logf(1.f + __expf(v)); }
// A-fragment K index for 16-bit 16x32 A layout (ISA 7.12.2)
DEV int kA(int j, int h) { return ((j & 8) << 1) | (j & 7) | (h << 3); }

// ---------------- workspace layout (bytes, 256-aligned) ----------------
constexpr size_t OFF_NM  = 0;         // nmeans:  8*1024 f32      = 32768
constexpr size_t OFF_CP  = 32768;     // cp:      8*16 f32        = 512
constexpr size_t OFF_BK  = 33280;     // buckets: 8*2048 i32      = 65536
constexpr size_t OFF_SI  = 98816;     // sidx:    8*2048 i32      = 65536
constexpr size_t OFF_WF  = 164352;    // Wfrag:   4*32*32*16 bf16 = 131072
constexpr size_t OFF_DWF = 295424;    // dtWfrag: 64*32*16 bf16   = 65536
constexpr size_t OFF_XD  = 360960;    // x_dbl:   8*64*2048 f32   = 4194304
constexpr size_t OFF_BC  = 4555264;   // BC:      8*2048*32 f32   = 2097152

// ---------------- prep: normalized means ----------------
__global__ void k_norm_means(const float* __restrict__ means, float* __restrict__ nmeans) {
    __shared__ float red[256];
    int k = blockIdx.x, t = threadIdx.x;
    float ss = 0.f;
    for (int d = t; d < DD; d += 256) { float v = means[k * DD + d]; ss += v * v; }
    red[t] = ss; __syncthreads();
    for (int s = 128; s; s >>= 1) { if (t < s) red[t] += red[t + s]; __syncthreads(); }
    float inv = 1.f / fmaxf(sqrtf(red[0]), 1e-12f);
    for (int d = t; d < DD; d += 256) nmeans[k * DD + d] = means[k * DD + d] * inv;
}

// ---------------- prep: cluster prompts cp[k][n] = means[k] . pw[n] ----------------
__global__ void k_cp(const float* __restrict__ means, const float* __restrict__ pw,
                     float* __restrict__ cp) {
    int i = threadIdx.x;
    if (i >= KC * NS) return;
    int k = i >> 4, n = i & 15;
    float acc = 0.f;
    for (int d = 0; d < DD; ++d) acc += means[k * DD + d] * pw[n * DD + d];
    cp[i] = acc;
}

// ---------------- prep: x_proj weight into B-fragment layout ----------------
// Wfrag[ct][ki][lane][j] = bf16(W[(ct*16 + n) * 1024 + ki*32 + 16h + j])
__global__ void k_wfrag(const float* __restrict__ W, unsigned short* __restrict__ wf) {
    int flat = blockIdx.x * blockDim.x + threadIdx.x;   // 65536 total
    int j = flat & 15, lane = (flat >> 4) & 31, ki = (flat >> 9) & 31, ct = flat >> 14;
    int n = lane & 15, h = lane >> 4;
    wf[flat] = f2bf(W[(ct * 16 + n) * DD + ki * 32 + (h << 4) + j]);
}

// ---------------- prep: dt_proj weight into A-fragment layout ----------------
// dtWfrag[t][lane][j] = bf16(dtW[(t*16 + m)*32 + kA(j,h)])
__global__ void k_dtwfrag(const float* __restrict__ dtW, unsigned short* __restrict__ dwf) {
    int flat = blockIdx.x * blockDim.x + threadIdx.x;   // 32768 total
    int j = flat & 15, lane = (flat >> 4) & 31, t = flat >> 9;
    int m = lane & 15, h = lane >> 4;
    dwf[flat] = f2bf(dtW[(t * 16 + m) * RK + kA(j, h)]);
}

// ---------------- bucketing: cosine argmax ----------------
__global__ void k_buckets(const float* __restrict__ x, const float* __restrict__ nmeans,
                          int* __restrict__ buckets) {
    __shared__ float sh[9 * 128];
    int row = blockIdx.x;      // b*NN + n
    int t = threadIdx.x;
    const float* xr = x + (size_t)row * DD;
    float ss = 0.f, dot[8] = {0, 0, 0, 0, 0, 0, 0, 0};
    for (int d = t; d < DD; d += 128) {
        float v = xr[d];
        ss += v * v;
#pragma unroll
        for (int k = 0; k < 8; ++k) dot[k] += v * nmeans[k * DD + d];
    }
    sh[t] = ss;
#pragma unroll
    for (int k = 0; k < 8; ++k) sh[(k + 1) * 128 + t] = dot[k];
    __syncthreads();
    for (int s = 64; s; s >>= 1) {
        if (t < s) {
#pragma unroll
            for (int q = 0; q < 9; ++q) sh[q * 128 + t] += sh[q * 128 + t + s];
        }
        __syncthreads();
    }
    if (t == 0) {
        float inv = 1.f / fmaxf(sqrtf(sh[0]), 1e-12f);
        int best = 0; float bv = -INFINITY;
        for (int k = 0; k < 8; ++k) {
            float sc = sh[(k + 1) * 128] * inv;
            if (sc > bv) { bv = sc; best = k; }
        }
        buckets[row] = best;
    }
}

// ---------------- stable counting sort per batch ----------------
__global__ void k_sort(const int* __restrict__ buckets, int* __restrict__ sidx) {
    __shared__ int cnt[256 * 8];
    __shared__ int base[8];
    int b = blockIdx.x, t = threadIdx.x;
    const int* bk = buckets + b * NN;
    int loc[8] = {0, 0, 0, 0, 0, 0, 0, 0};
    int n0 = t * 8;
    for (int e = 0; e < 8; ++e) loc[bk[n0 + e]]++;
#pragma unroll
    for (int k = 0; k < 8; ++k) cnt[t * 8 + k] = loc[k];
    __syncthreads();
    if (t < 8) {             // exclusive prefix over threads for bucket t
        int run = 0;
        for (int i = 0; i < 256; ++i) { int c = cnt[i * 8 + t]; cnt[i * 8 + t] = run; run += c; }
        base[t] = run;       // total count
    }
    __syncthreads();
    if (t == 0) { int run = 0; for (int k = 0; k < 8; ++k) { int c = base[k]; base[k] = run; run += c; } }
    __syncthreads();
#pragma unroll
    for (int k = 0; k < 8; ++k) loc[k] = base[k] + cnt[t * 8 + k];
    for (int e = 0; e < 8; ++e) { int k = bk[n0 + e]; sidx[b * NN + loc[k]] = n0 + e; loc[k]++; }
}

// ---------------- x_dbl GEMM: [16 l x 64 c] per wave via WMMA bf16 ----------------
__global__ void __launch_bounds__(256) k_xdbl(const float* __restrict__ x,
                                              const int* __restrict__ sidx,
                                              const unsigned short* __restrict__ wf,
                                              float* __restrict__ xdbl) {
    __shared__ unsigned short asA[8][16][32];
    int blk = blockIdx.x;            // 128 blocks: b = blk>>4, lblock = blk&15
    int b = blk >> 4, lb = blk & 15;
    int tid = threadIdx.x, w = tid >> 5, lane = tid & 31;
    int l0 = lb * 128 + w * 16;
    int m = lane & 15, h = lane >> 4;

    v8f zero = {0.f, 0.f, 0.f, 0.f, 0.f, 0.f, 0.f, 0.f};
    v8f acc[4] = {zero, zero, zero, zero};

    int grow = sidx[b * NN + l0 + m];
    const float* srcbase = x + (size_t)(b * NN + grow) * DD + h * 16;

    for (int ki = 0; ki < 32; ++ki) {
        // gather + convert A tile (16 l-rows x 32 k) into LDS
        const float4* s4 = reinterpret_cast<const float4*>(srcbase + ki * 32);
#pragma unroll
        for (int q = 0; q < 4; ++q) {
            float4 v = s4[q];
            int c = h * 16 + q * 4;
            asA[w][m][c + 0] = f2bf(v.x); asA[w][m][c + 1] = f2bf(v.y);
            asA[w][m][c + 2] = f2bf(v.z); asA[w][m][c + 3] = f2bf(v.w);
        }
        __syncthreads();
        U16 a;
#pragma unroll
        for (int j = 0; j < 16; ++j) a.s[j] = asA[w][m][h * 16 + j];
#pragma unroll
        for (int ct = 0; ct < 4; ++ct) {
            U16 bf;
            const U128* p = reinterpret_cast<const U128*>(wf + ((size_t)((ct * 32 + ki) * 32 + lane)) * 16);
            bf.q[0] = p[0]; bf.q[1] = p[1];
            acc[ct] = __builtin_amdgcn_wmma_f32_16x16x32_bf16(
                false, a.v, false, bf.v, (short)0, acc[ct], false, false);
        }
        __syncthreads();
    }
#pragma unroll
    for (int ct = 0; ct < 4; ++ct) {
        int c = ct * 16 + m;     // N dim = c
#pragma unroll
        for (int r = 0; r < 8; ++r) {
            xdbl[((size_t)b * CC + c) * NN + l0 + r + 8 * h] = acc[ct][r];
        }
    }
}

// ---------------- pack B/C(+prompt) into [b][l][32] ----------------
__global__ void k_bc(const float* __restrict__ xdbl, const float* __restrict__ cp,
                     const int* __restrict__ buckets, const int* __restrict__ sidx,
                     float* __restrict__ bc) {
    int i = blockIdx.x * 256 + threadIdx.x;     // BB*NN*32 total
    int n2 = i & 31, l = (i >> 5) & (NN - 1), b = i >> 16;
    int g = sidx[b * NN + l];
    float v;
    if (n2 < 16) {
        v = xdbl[((size_t)b * CC + 32 + n2) * NN + l];
    } else {
        int k = buckets[b * NN + g];
        v = xdbl[((size_t)b * CC + 48 + (n2 - 16)) * NN + l] + cp[k * 16 + (n2 - 16)];
    }
    bc[((size_t)(b * NN) + l) * 32 + n2] = v;
}

// ---------------- fused delta-GEMM (WMMA) + selective scan ----------------
// one wave per block; 32 channels per block; 256 blocks
__global__ void __launch_bounds__(32) k_scan(const float* __restrict__ x,
                                             const float* __restrict__ xdbl,
                                             const unsigned short* __restrict__ dwf,
                                             const float* __restrict__ bc,
                                             const int* __restrict__ sidx,
                                             const float* __restrict__ A_logs,
                                             const float* __restrict__ Dvec,
                                             const float* __restrict__ dt_bias,
                                             float* __restrict__ out) {
    __shared__ float dS[32][17];       // delta staging (padded, conflict-free)
    __shared__ float bcS[16][32];
    __shared__ float biasS[32];
    __shared__ int   gS[16];

    int blk = blockIdx.x;
    int b = blk >> 5, d0 = (blk & 31) * 32;
    int lane = threadIdx.x;
    int n = lane & 15, h = lane >> 4;
    int d = d0 + lane;

    // exp(x) = exp2(x * log2e); bake log2e into A so inner loop uses raw v_exp_f32
    constexpr float LOG2E = 1.4426950408889634f;
    float Arow[16], hst[16];
#pragma unroll
    for (int q = 0; q < 16; ++q) { Arow[q] = -expf(A_logs[d * NS + q]) * LOG2E; hst[q] = 0.f; }
    float Dd = Dvec[d];
    biasS[lane] = dt_bias[d];

    // dtW A-fragments, resident in registers for the whole kernel
    int t0 = d0 >> 4;
    U16 a0, a1;
    { const U128* p = reinterpret_cast<const U128*>(dwf + ((size_t)t0 * 32 + lane) * 16);
      a0.q[0] = p[0]; a0.q[1] = p[1]; }
    { const U128* p = reinterpret_cast<const U128*>(dwf + ((size_t)(t0 + 1) * 32 + lane) * 16);
      a1.q[0] = p[0]; a1.q[1] = p[1]; }
    __syncthreads();

    const float* dts = xdbl + (size_t)b * CC * NN;   // rows 0..31 are dts
    v8f zero = {0.f, 0.f, 0.f, 0.f, 0.f, 0.f, 0.f, 0.f};

    for (int chunk = 0; chunk < NN / 16; ++chunk) {
        int l0 = chunk * 16;
        // B fragment: dts[r = kA(j,h)][l0+n] -> bf16
        U16 bbf;
#pragma unroll
        for (int j = 0; j < 16; ++j)
            bbf.s[j] = f2bf(dts[(size_t)kA(j, h) * NN + l0 + n]);

        v8f acc0 = zero, acc1 = zero;
        acc0 = __builtin_amdgcn_wmma_f32_16x16x32_bf16(false, a0.v, false, bbf.v, (short)0, acc0, false, false);
        acc1 = __builtin_amdgcn_wmma_f32_16x16x32_bf16(false, a1.v, false, bbf.v, (short)0, acc1, false, false);

        __syncthreads();   // prior chunk fully consumed (NOP for 1-wave WG)
#pragma unroll
        for (int r = 0; r < 8; ++r) {
            int dl = r + 8 * h;
            dS[dl][n]      = softplusf(acc0[r] + biasS[dl]);
            dS[dl + 16][n] = softplusf(acc1[r] + biasS[dl + 16]);
        }
#pragma unroll
        for (int rep = 0; rep < 16; ++rep) {
            int idx = rep * 32 + lane;
            int s = idx >> 5, c = idx & 31;
            bcS[s][c] = bc[((size_t)(b * NN) + l0 + s) * 32 + c];
        }
        if (lane < 16) gS[lane] = sidx[b * NN + l0 + lane];
        __syncthreads();

        for (int s = 0; s < 16; ++s) {
            float dt = dS[lane][s];
            int g = gS[s];
            float u = x[((size_t)(b * NN + g)) * DD + d];
            float du = dt * u;
            // broadcast B/C for this step as vectorized LDS loads (ds_load_b128 x8)
            const float4* bv4 = reinterpret_cast<const float4*>(&bcS[s][0]);
            float Bv[16], Cv[16];
#pragma unroll
            for (int q4 = 0; q4 < 4; ++q4) {
                float4 bq = bv4[q4];
                float4 cq = bv4[q4 + 4];
                Bv[q4 * 4 + 0] = bq.x; Bv[q4 * 4 + 1] = bq.y; Bv[q4 * 4 + 2] = bq.z; Bv[q4 * 4 + 3] = bq.w;
                Cv[q4 * 4 + 0] = cq.x; Cv[q4 * 4 + 1] = cq.y; Cv[q4 * 4 + 2] = cq.z; Cv[q4 * 4 + 3] = cq.w;
            }
            float y = 0.f;
#pragma unroll
            for (int q = 0; q < 16; ++q) {
                float dA = __builtin_amdgcn_exp2f(dt * Arow[q]);   // raw v_exp_f32
                hst[q] = dA * hst[q] + du * Bv[q];
                y += hst[q] * Cv[q];
            }
            out[((size_t)(b * NN + g)) * DD + d] = y + Dd * u;   // fused unsort scatter
        }
    }
}

// ---------------- host launcher ----------------
extern "C" void kernel_launch(void* const* d_in, const int* in_sizes, int n_in,
                              void* d_out, int out_size, void* d_ws, size_t ws_size,
                              hipStream_t stream) {
    (void)in_sizes; (void)n_in; (void)out_size; (void)ws_size;
    const float* x     = (const float*)d_in[0];
    const float* means = (const float*)d_in[1];
    const float* pw    = (const float*)d_in[2];
    const float* xpw   = (const float*)d_in[3];
    const float* dtw   = (const float*)d_in[4];
    const float* dtb   = (const float*)d_in[5];
    const float* alog  = (const float*)d_in[6];
    const float* Dvec  = (const float*)d_in[7];
    float* out = (float*)d_out;

    char* ws = (char*)d_ws;
    float*          nmeans  = (float*)(ws + OFF_NM);
    float*          cp      = (float*)(ws + OFF_CP);
    int*            buckets = (int*)(ws + OFF_BK);
    int*            sidx    = (int*)(ws + OFF_SI);
    unsigned short* wfrag   = (unsigned short*)(ws + OFF_WF);
    unsigned short* dwf     = (unsigned short*)(ws + OFF_DWF);
    float*          xdbl    = (float*)(ws + OFF_XD);
    float*          bcbuf   = (float*)(ws + OFF_BC);

    k_norm_means<<<KC, 256, 0, stream>>>(means, nmeans);
    k_cp<<<1, 128, 0, stream>>>(means, pw, cp);
    k_wfrag<<<256, 256, 0, stream>>>(xpw, wfrag);
    k_dtwfrag<<<128, 256, 0, stream>>>(dtw, dwf);
    k_buckets<<<BB * NN, 128, 0, stream>>>(x, nmeans, buckets);
    k_sort<<<BB, 256, 0, stream>>>(buckets, sidx);
    k_xdbl<<<128, 256, 0, stream>>>(x, sidx, wfrag, xdbl);
    k_bc<<<BB * NN * 32 / 256, 256, 0, stream>>>(xdbl, cp, buckets, sidx, bcbuf);
    k_scan<<<BB * 32, 32, 0, stream>>>(x, xdbl, dwf, bcbuf, sidx, alog, Dvec, dtb, out);
}
